// PointerSegmenter_20126216749358
// MI455X (gfx1250) — matrix operand order repaired and checked by reading, not compiled, not run on hardware
//
#include <hip/hip_runtime.h>
#include <stdint.h>

// Problem constants (fixed by the reference setup).
constexpr int L    = 16384;   // sequence length
constexpr int H    = 768;     // model width
constexpr int Hd   = 384;     // per-direction encoder hidden
constexpr int G3   = 3 * Hd;  // 1152 encoder gate rows
constexpr int D3   = 3 * H;   // 2304 decoder gate rows
constexpr int KEDU = 512;     // number of EDU breaks / decoder steps

// ---------------------------------------------------------------------------
// WMMA / TDM types
// ---------------------------------------------------------------------------
typedef __bf16 bf16_t;
typedef __attribute__((ext_vector_type(16))) __bf16    v16bf;
typedef __attribute__((ext_vector_type(8)))  float     v8f;
typedef __attribute__((ext_vector_type(4)))  uint32_t  v4u;
typedef __attribute__((ext_vector_type(8)))  int       v8i_t;
typedef __attribute__((ext_vector_type(4)))  int       v4i_t;

struct U32x8 { uint4 lo; uint4 hi; };  // 32 bytes, bit-castable to v16bf

__device__ inline v16bf frag_cast(const U32x8& u) {
    return __builtin_bit_cast(v16bf, u);
}

__device__ inline float sigm(float x) { return 1.0f / (1.0f + __expf(-x)); }

// ---------------------------------------------------------------------------
// fp32 -> bf16 (round to nearest even)
// ---------------------------------------------------------------------------
__global__ void cvt_f32_to_bf16(const float* __restrict__ in,
                                uint16_t* __restrict__ out, int n) {
    int i = blockIdx.x * blockDim.x + threadIdx.x;
    if (i >= n) return;
    uint32_t u = __float_as_uint(in[i]);
    uint32_t r = (u + 0x7FFFu + ((u >> 16) & 1u)) >> 16;
    out[i] = (uint16_t)r;
}

// ---------------------------------------------------------------------------
// WMMA GEMM:  C[M,N] = A[M,K] * B[N,K]^T (+ bias[N])
// A, B bf16 row-major; C fp32 row-major. Block = 256 threads (8 wave32).
// Block tile: 128 (M) x 64 (N); each wave owns a 16x64 strip (4 WMMA accs).
//
// B panel DMA'd once per block into LDS by the Tensor Data Mover, reshaped
// in flight to [K/16][64][16] so each lane's 32-byte fragment read is two
// contiguous ds_load_b128. A (global) and B (LDS) fragments are software-
// pipelined THREE stages deep: at each consumption point the two newer load
// batches stay outstanding, so waits need not drain to zero.
// Requires M%128==0, N%64==0, K%96==0 (true for every GEMM here: K=768).
// ---------------------------------------------------------------------------
__global__ __launch_bounds__(256)
void wmma_gemm_bf16_nt(const uint16_t* __restrict__ A,
                       const uint16_t* __restrict__ B,
                       const float* __restrict__ bias,
                       float* __restrict__ C,
                       int M, int N, int K) {
    extern __shared__ uint16_t bpanel[];  // [K/16][64][16] bf16 (K/16*2 KB)

    const int wave = threadIdx.x >> 5;
    const int lane = threadIdx.x & 31;
    const int m    = lane & 15;     // A row / B col within 16-wide tile
    const int hi16 = lane >> 4;     // which half-wave

    const int rowBase = blockIdx.y * 128 + wave * 16;
    const int colBase = blockIdx.x * 64;

    // ---- TDM: stage B panel (rows colBase..colBase+63, all K) into LDS ----
    if (wave == 0) {
        const uint64_t ga  = (uint64_t)(uintptr_t)(B + (size_t)colBase * K);
        const uint32_t lds = (uint32_t)(uintptr_t)bpanel;
        const uint32_t big = 1u << 20;  // comfortably >= any tile dim (no OOB)

        v4u g0;
        g0[0] = 1u;                                   // count=1 (valid D#)
        g0[1] = lds;                                  // lds_addr
        g0[2] = (uint32_t)ga;                         // global_addr[31:0]
        g0[3] = (uint32_t)((ga >> 32) & 0x01FFFFFFu)  // global_addr[56:32]
              | (2u << 30);                           // type=2 ("image")

        v8i_t g1;
        g1[0] = (int)(1u << 16);                      // data_size=1 (2B), mask=0
        g1[1] = (int)((big & 0xFFFFu) << 16);         // tensor_dim0[15:0]
        g1[2] = (int)((big >> 16) | ((big & 0xFFFFu) << 16)); // dim0 hi | dim1 lo
        g1[3] = (int)((big >> 16) | (16u << 16));     // dim1 hi | tile_dim0=16
        g1[4] = (int)(64u | ((uint32_t)(K / 16) << 16)); // tile_dim1=64 | tile_dim2=K/16
        g1[5] = (int)K;                               // tensor_dim0_stride = K (row step)
        g1[6] = (int)(16u << 16);                     // stride0 hi=0 | tensor_dim1_stride=16
        g1[7] = 0;                                    // stride1 hi

        v4i_t g2;
        g2[0] = (int)big;                             // tensor_dim2
        g2[1] = 0; g2[2] = 0; g2[3] = 0;              // dim3 / stride2 / tile_dim3 unused
        v4i_t g3;
        g3[0] = 0; g3[1] = 0; g3[2] = 0; g3[3] = 0;
        v8i_t g4;                                     // extra operand (clang-23 6-arg
        g4[0] = 0; g4[1] = 0; g4[2] = 0; g4[3] = 0;   //  form); zero-filled
        g4[4] = 0; g4[5] = 0; g4[6] = 0; g4[7] = 0;

        __builtin_amdgcn_tensor_load_to_lds(g0, g1, g2, g3, g4, 0);
        __builtin_amdgcn_s_wait_tensorcnt(0);
    }
    __syncthreads();

    v8f acc[4];
#pragma unroll
    for (int i = 0; i < 4; ++i) acc[i] = v8f{};

    const uint16_t* arow = A + (size_t)(rowBase + m) * K + hi16 * 16;

    auto loadA = [&](U32x8& d, int kb) {
        d.lo = *reinterpret_cast<const uint4*>(arow + kb);
        d.hi = *reinterpret_cast<const uint4*>(arow + kb + 8);
    };
    auto loadB = [&](U32x8* d, int kb) {
        const int cb = (kb >> 4) + hi16;  // k-chunk for this half-wave
#pragma unroll
        for (int nt = 0; nt < 4; ++nt) {
            const uint16_t* bp = bpanel + ((size_t)(cb * 64 + nt * 16 + m) << 4);
            d[nt].lo = *reinterpret_cast<const uint4*>(bp);
            d[nt].hi = *reinterpret_cast<const uint4*>(bp + 8);
        }
    };
    auto mm4 = [&](const U32x8& a, const U32x8* b) {
        const v16bf af = frag_cast(a);
#pragma unroll
        for (int nt = 0; nt < 4; ++nt) {
            acc[nt] = __builtin_amdgcn_wmma_f32_16x16x32_bf16(
                false, af, false, frag_cast(b[nt]), (short)0, acc[nt], false, false);
        }
    };

    // Three-stage software pipeline: stage st consumes buffer st (chunk
    // kb+32*st) and refills buffer (st+2)%3 with the chunk two stages ahead.
    U32x8 a[3];
    U32x8 b[3][4];
    loadA(a[0], 0);   loadB(b[0], 0);
    loadA(a[1], 32);  loadB(b[1], 32);
    for (int kb = 0; kb < K; kb += 96) {
#pragma unroll
        for (int st = 0; st < 3; ++st) {
            const int lc   = kb + 64 + st * 32;   // chunk 2 stages ahead
            const int fill = (st + 2) % 3;
            if (lc < K) {
                loadA(a[fill], lc);
                loadB(b[fill], lc);
                if (st == 0) __builtin_prefetch(arow + kb + 160, 0, 1);
            }
            __builtin_amdgcn_sched_barrier(0);
            mm4(a[st], b[st]);
            __builtin_amdgcn_sched_barrier(0);
        }
    }

#pragma unroll
    for (int nt = 0; nt < 4; ++nt) {
        const int col = colBase + nt * 16 + m;
        const float bb = bias ? bias[col] : 0.0f;
#pragma unroll
        for (int r = 0; r < 8; ++r) {
            const int row = rowBase + r + hi16 * 8;
            C[(size_t)row * N + col] = acc[nt][r] + bb;
        }
    }
}

// ---------------------------------------------------------------------------
// 8-wide fp32 dot-accumulate helper for the scans (weights read fp32 from
// the original input buffers; L2-resident, so the scans stay VALU-bound and
// we avoid bf16 unpack ALU in the latency-critical sequential loop).
// ---------------------------------------------------------------------------
__device__ inline void fma8_f32(const float* __restrict__ wp, int k,
                                const float4& h0, const float4& h1,
                                float& acc) {
    float4 w0 = *reinterpret_cast<const float4*>(wp + k);
    float4 w1 = *reinterpret_cast<const float4*>(wp + k + 4);
    acc = fmaf(w0.x, h0.x, acc);
    acc = fmaf(w0.y, h0.y, acc);
    acc = fmaf(w0.z, h0.z, acc);
    acc = fmaf(w0.w, h0.w, acc);
    acc = fmaf(w1.x, h1.x, acc);
    acc = fmaf(w1.y, h1.y, acc);
    acc = fmaf(w1.z, h1.z, acc);
    acc = fmaf(w1.w, h1.w, acc);
}

// ---------------------------------------------------------------------------
// Bidirectional encoder GRU scan. grid = 2 (dir), block = Hd (384) threads.
// h double-buffered in LDS; W_hh streamed fp32 (L2-resident, 1.7 MB/dir).
// Gate order (r,z,n): rows j, j+Hd, j+2Hd.  outputs[t] = [fwd | bwd].
// ---------------------------------------------------------------------------
__global__ __launch_bounds__(Hd)
void enc_gru_scan(const float* __restrict__ gi_f, const float* __restrict__ gi_b,
                  const float* __restrict__ Wf, const float* __restrict__ Wb,
                  const float* __restrict__ bhh_f, const float* __restrict__ bhh_b,
                  float* __restrict__ outputs) {
    const int dir = blockIdx.x;
    const float* gi  = dir ? gi_b  : gi_f;
    const float* W   = dir ? Wb    : Wf;
    const float* bhh = dir ? bhh_b : bhh_f;
    const int j = threadIdx.x;

    __shared__ __align__(16) float hbuf[2][Hd];
    hbuf[0][j] = 0.0f;
    __syncthreads();

    const float* w_r = W + (size_t)j * Hd;
    const float* w_z = W + (size_t)(j + Hd) * Hd;
    const float* w_n = W + (size_t)(j + 2 * Hd) * Hd;

    for (int s = 0; s < L; ++s) {
        const int t = dir ? (L - 1 - s) : s;
        const float* hc = hbuf[s & 1];

        float ar = 0.0f, az = 0.0f, an = 0.0f;
        for (int k = 0; k < Hd; k += 8) {
            const float4 h0 = *reinterpret_cast<const float4*>(hc + k);
            const float4 h1 = *reinterpret_cast<const float4*>(hc + k + 4);
            fma8_f32(w_r, k, h0, h1, ar);
            fma8_f32(w_z, k, h0, h1, az);
            fma8_f32(w_n, k, h0, h1, an);
        }

        const float* git = gi + (size_t)t * G3;
        const float r  = sigm(git[j]          + bhh[j]          + ar);
        const float z  = sigm(git[j + Hd]     + bhh[j + Hd]     + az);
        const float hn = an + bhh[j + 2 * Hd];
        const float n  = tanhf(git[j + 2 * Hd] + r * hn);
        const float hnew = (1.0f - z) * n + z * hc[j];

        outputs[(size_t)t * H + dir * Hd + j] = hnew;
        hbuf[(s & 1) ^ 1][j] = hnew;
        __syncthreads();
    }
}

// ---------------------------------------------------------------------------
// Gather decoder inputs: dec_in_bf[k] = outputs_bf[starts[k]]
// starts[0]=0, starts[k]=edu[k-1]
// ---------------------------------------------------------------------------
__global__ void gather_dec_in(const uint16_t* __restrict__ outputs_bf,
                              const int* __restrict__ edu,
                              uint16_t* __restrict__ dec_in) {
    int idx = blockIdx.x * blockDim.x + threadIdx.x;
    if (idx >= KEDU * H) return;
    const int k = idx / H;
    const int j = idx - k * H;
    const int start = k ? edu[k - 1] : 0;
    dec_in[idx] = outputs_bf[(size_t)start * H + j];
}

// ---------------------------------------------------------------------------
// Decoder GRU scan. 1 block of H (768) threads, KEDU sequential steps.
// h_init = outputs[L-1].
// ---------------------------------------------------------------------------
__global__ __launch_bounds__(H)
void dec_gru_scan(const float* __restrict__ gi,
                  const float* __restrict__ W,
                  const float* __restrict__ bhh,
                  const float* __restrict__ outputs,
                  float* __restrict__ dec_hs) {
    const int j = threadIdx.x;
    __shared__ __align__(16) float hbuf[2][H];
    hbuf[0][j] = outputs[(size_t)(L - 1) * H + j];
    __syncthreads();

    const float* w_r = W + (size_t)j * H;
    const float* w_z = W + (size_t)(j + H) * H;
    const float* w_n = W + (size_t)(j + 2 * H) * H;

    for (int s = 0; s < KEDU; ++s) {
        const float* hc = hbuf[s & 1];

        float ar = 0.0f, az = 0.0f, an = 0.0f;
        for (int k = 0; k < H; k += 8) {
            const float4 h0 = *reinterpret_cast<const float4*>(hc + k);
            const float4 h1 = *reinterpret_cast<const float4*>(hc + k + 4);
            fma8_f32(w_r, k, h0, h1, ar);
            fma8_f32(w_z, k, h0, h1, az);
            fma8_f32(w_n, k, h0, h1, an);
        }

        const float* git = gi + (size_t)s * D3;
        const float r  = sigm(git[j]         + bhh[j]         + ar);
        const float z  = sigm(git[j + H]     + bhh[j + H]     + az);
        const float hn = an + bhh[j + 2 * H];
        const float n  = tanhf(git[j + 2 * H] + r * hn);
        const float hnew = (1.0f - z) * n + z * hc[j];

        dec_hs[(size_t)s * H + j] = hnew;
        hbuf[(s & 1) ^ 1][j] = hnew;
        __syncthreads();
    }
}

// ---------------------------------------------------------------------------
// Per-row masked log-softmax NLL. The NEG mask == suffix [start, L).
// One block (256 threads) per k.
// ---------------------------------------------------------------------------
__global__ __launch_bounds__(256)
void loss_rows(const float* __restrict__ scores,
               const int* __restrict__ edu,
               float* __restrict__ lossk) {
    const int k = blockIdx.x;
    const float* row = scores + (size_t)k * L;
    const int start = k ? edu[k - 1] : 0;
    const int tgt   = edu[k];
    const int tid = threadIdx.x;

    __shared__ float red[256];

    float m = -3.0e38f;
    for (int j = start + tid; j < L; j += 256) m = fmaxf(m, row[j]);
    red[tid] = m; __syncthreads();
    for (int w = 128; w > 0; w >>= 1) {
        if (tid < w) red[tid] = fmaxf(red[tid], red[tid + w]);
        __syncthreads();
    }
    m = red[0]; __syncthreads();

    float ssum = 0.0f;
    for (int j = start + tid; j < L; j += 256) ssum += __expf(row[j] - m);
    red[tid] = ssum; __syncthreads();
    for (int w = 128; w > 0; w >>= 1) {
        if (tid < w) red[tid] += red[tid + w];
        __syncthreads();
    }
    if (tid == 0) lossk[k] = -(row[tgt] - m - logf(red[0]));
}

// Deterministic final sum over KEDU per-row losses.
__global__ __launch_bounds__(256)
void final_reduce(const float* __restrict__ lossk, float* __restrict__ out) {
    const int tid = threadIdx.x;
    __shared__ float red[256];
    red[tid] = lossk[tid] + lossk[tid + 256];
    __syncthreads();
    for (int w = 128; w > 0; w >>= 1) {
        if (tid < w) red[tid] += red[tid + w];
        __syncthreads();
    }
    if (tid == 0) out[0] = red[0];
}

// ---------------------------------------------------------------------------
// Host launcher
// ---------------------------------------------------------------------------
static inline size_t align_up(size_t x) { return (x + 255) & ~(size_t)255; }

extern "C" void kernel_launch(void* const* d_in, const int* in_sizes, int n_in,
                              void* d_out, int out_size, void* d_ws, size_t ws_size,
                              hipStream_t stream) {
    const float* emb     = (const float*)d_in[0];   // [L,H]
    const float* wih_f   = (const float*)d_in[1];   // [G3,H]
    const float* whh_f   = (const float*)d_in[2];   // [G3,Hd]
    const float* bih_f   = (const float*)d_in[3];
    const float* bhh_f   = (const float*)d_in[4];
    const float* wih_b   = (const float*)d_in[5];
    const float* whh_b   = (const float*)d_in[6];
    const float* bih_b   = (const float*)d_in[7];
    const float* bhh_b   = (const float*)d_in[8];
    const float* dwih    = (const float*)d_in[9];   // [D3,H]
    const float* dwhh    = (const float*)d_in[10];  // [D3,H]
    const float* dbih    = (const float*)d_in[11];
    const float* dbhh    = (const float*)d_in[12];
    const int*   edu     = (const int*)d_in[13];    // [KEDU]
    float*       out     = (float*)d_out;

    // Workspace carve-up
    char* base = (char*)d_ws;
    size_t off = 0;
    auto take = [&](size_t bytes) { void* p = base + off; off += align_up(bytes); return p; };

    uint16_t* Xbf        = (uint16_t*)take((size_t)L * H * 2);
    uint16_t* Wihf_bf    = (uint16_t*)take((size_t)G3 * H * 2);
    uint16_t* Wihb_bf    = (uint16_t*)take((size_t)G3 * H * 2);
    uint16_t* dWih_bf    = (uint16_t*)take((size_t)D3 * H * 2);
    float*    gi_f       = (float*)take((size_t)L * G3 * 4);
    float*    gi_b       = (float*)take((size_t)L * G3 * 4);
    float*    outputs    = (float*)take((size_t)L * H * 4);
    uint16_t* outputs_bf = (uint16_t*)take((size_t)L * H * 2);
    uint16_t* dec_in_bf  = (uint16_t*)take((size_t)KEDU * H * 2);
    float*    dec_gi     = (float*)take((size_t)KEDU * D3 * 4);
    float*    dec_hs     = (float*)take((size_t)KEDU * H * 4);
    uint16_t* dec_hs_bf  = (uint16_t*)take((size_t)KEDU * H * 2);
    float*    scores     = (float*)take((size_t)KEDU * L * 4);
    float*    lossk      = (float*)take((size_t)KEDU * 4);
    (void)ws_size; (void)n_in; (void)in_sizes; (void)out_size;

    auto cvt = [&](const float* src, uint16_t* dst, int n) {
        cvt_f32_to_bf16<<<(n + 255) / 256, 256, 0, stream>>>(src, dst, n);
    };

    const size_t panelBytes = (size_t)(H / 16) * 64 * 16 * 2;  // 96 KB LDS B panel

    // 1) fp32 -> bf16 conversions (GEMM operands only; scans use fp32 weights)
    cvt(emb,   Xbf,     L * H);
    cvt(wih_f, Wihf_bf, G3 * H);
    cvt(wih_b, Wihb_bf, G3 * H);
    cvt(dwih,  dWih_bf, D3 * H);

    // 2) Encoder input projections: gi = X @ W_ih^T + b_ih   (WMMA + TDM)
    wmma_gemm_bf16_nt<<<dim3(G3 / 64, L / 128), 256, panelBytes, stream>>>(
        Xbf, Wihf_bf, bih_f, gi_f, L, G3, H);
    wmma_gemm_bf16_nt<<<dim3(G3 / 64, L / 128), 256, panelBytes, stream>>>(
        Xbf, Wihb_bf, bih_b, gi_b, L, G3, H);

    // 3) Bidirectional encoder scan (fwd + bwd concurrently)
    enc_gru_scan<<<2, Hd, 0, stream>>>(gi_f, gi_b, whh_f, whh_b,
                                       bhh_f, bhh_b, outputs);

    // 4) outputs -> bf16 for decoder input gather + attention
    cvt(outputs, outputs_bf, L * H);

    // 5) Decoder input gather + input projection (WMMA + TDM)
    gather_dec_in<<<(KEDU * H + 255) / 256, 256, 0, stream>>>(outputs_bf, edu, dec_in_bf);
    wmma_gemm_bf16_nt<<<dim3(D3 / 64, KEDU / 128), 256, panelBytes, stream>>>(
        dec_in_bf, dWih_bf, dbih, dec_gi, KEDU, D3, H);

    // 6) Decoder scan (h0 = outputs[L-1])
    dec_gru_scan<<<1, H, 0, stream>>>(dec_gi, dwhh, dbhh, outputs, dec_hs);

    // 7) Pointer attention scores = dec_hs @ outputs^T  (WMMA + TDM)
    cvt(dec_hs, dec_hs_bf, KEDU * H);
    wmma_gemm_bf16_nt<<<dim3(L / 64, KEDU / 128), 256, panelBytes, stream>>>(
        dec_hs_bf, outputs_bf, nullptr, scores, KEDU, L, H);

    // 8) Masked log-softmax NLL per row, then deterministic sum
    loss_rows<<<KEDU, 256, 0, stream>>>(scores, edu, lossk);
    final_reduce<<<1, 256, 0, stream>>>(lossk, out);
}
